// PKM_59871844106955
// MI455X (gfx1250) — matrix operand order repaired and checked by reading, not compiled, not run on hardware
//
#include <hip/hip_runtime.h>
#include <hip/hip_bf16.h>
#include <stdint.h>

#define DIM      1024
#define HEADS    4
#define NUM_KEYS 256
#define TOPK     32
#define DIM_HEAD 256
#define DH2      128     // DIM_HEAD/2
#define DQ       1024    // DIM_HEAD*HEADS
#define BT       2048    // B*T
#define BN_EPS   1e-5f

typedef __attribute__((ext_vector_type(16))) _Float16 v16h;
typedef __attribute__((ext_vector_type(8)))  float    v8f;

union FragH {
    v16h     v;
    _Float16 h[16];
    uint32_t u[8];
    uint4    q[2];
};

// A K index used here for VGPR v (0..7), lane half (0/1); contiguous k = 16*half + 2v
__device__ __forceinline__ int kOfA(int v, int half) {
    return 2 * v + 16 * half;
}
// B K index used here, contiguous as in A (lanes 0-15: K=0..15 across V0..7; lanes 16-31: K=16..31)
__device__ __forceinline__ int kOfB(int v, int half) {
    return 2 * v + 16 * half;
}

// Flat shared pointers carry the LDS byte offset in addr[31:0] (ISA 10.2 LDS aperture).
__device__ __forceinline__ uint32_t lds_u32(const void* p) {
    return (uint32_t)(uintptr_t)p;
}

// CDNA5 async copy global -> LDS, 16B per lane, tracked by ASYNCcnt.
__device__ __forceinline__ void async_ld16(uint32_t lds, const void* g) {
    asm volatile("global_load_async_to_lds_b128 %0, %1, off"
                 :: "v"(lds), "v"(g) : "memory");
}
__device__ __forceinline__ void wait_async_le6() { asm volatile("s_wait_asynccnt 0x6" ::: "memory"); }
__device__ __forceinline__ void wait_async_le0() { asm volatile("s_wait_asynccnt 0x0" ::: "memory"); }

// ---------------- f32 -> f16 convert (elementwise) ----------------
__global__ void conv_f16_kernel(const float* __restrict__ src, _Float16* __restrict__ dst, int n) {
    int base = (blockIdx.x * 256 + threadIdx.x) * 4;
    if (base + 3 < n) {
        const float4 f = *reinterpret_cast<const float4*>(src + base);
        dst[base + 0] = (_Float16)f.x;
        dst[base + 1] = (_Float16)f.y;
        dst[base + 2] = (_Float16)f.z;
        dst[base + 3] = (_Float16)f.w;
    }
}

// ---------------- W_q (k,n) -> wqt (n,k) f16, LDS tiled transpose ----------------
__global__ void transpose_wq_kernel(const float* __restrict__ W, _Float16* __restrict__ wqt) {
    __shared__ float tile[32][33];
    const int bx = blockIdx.x, by = blockIdx.y;
    const int tx = threadIdx.x, ty = threadIdx.y;
#pragma unroll
    for (int i = 0; i < 32; i += 8) {
        int k = by * 32 + ty + i;
        int n = bx * 32 + tx;
        tile[ty + i][tx] = W[(size_t)k * DQ + n];
    }
    __syncthreads();
#pragma unroll
    for (int i = 0; i < 32; i += 8) {
        int n = bx * 32 + ty + i;
        int k = by * 32 + tx;
        wqt[(size_t)n * DIM + k] = (_Float16)tile[tx][ty + i];
    }
}

// ---------------- q = x @ W_q via async-LDS double-buffered WMMA ----------------
// Block tile 128(M) x 64(N), K-step 64, 8 waves: wave w owns rows [w*16, w*16+16),
// all 64 N columns (4 accumulators). Operand tiles staged with
// GLOBAL_LOAD_ASYNC_TO_LDS_B128 (6 async ops / wave / tile), double buffered.
#define KS 64
__global__ void __launch_bounds__(256) gemm_q_kernel(const _Float16* __restrict__ xh,
                                                     const _Float16* __restrict__ wqt,
                                                     float* __restrict__ q) {
    const int tid = threadIdx.x, wave = tid >> 5, lane = tid & 31;
    const int m0 = blockIdx.y * 128;
    const int n0 = blockIdx.x * 64;
    const int row = lane & 15, half = lane >> 4;

    __shared__ _Float16 a_lds[2][128][KS];   // 32 KB
    __shared__ _Float16 b_lds[2][64][KS];    // 16 KB

    // ---- async tile fill: A = 1024 chunks of 16B, B = 512 chunks ----
    auto issue_tile = [&](int t, int buf) {
        const int k0 = t * KS;
#pragma unroll
        for (int i = 0; i < 4; ++i) {
            int c = tid + i * 256;               // 0..1023
            int r = c >> 3, cc = (c & 7) * 8;
            async_ld16(lds_u32(&a_lds[buf][r][cc]),
                       xh + (size_t)(m0 + r) * DIM + k0 + cc);
        }
#pragma unroll
        for (int i = 0; i < 2; ++i) {
            int c = tid + i * 256;               // 0..511
            int r = c >> 3, cc = (c & 7) * 8;
            async_ld16(lds_u32(&b_lds[buf][r][cc]),
                       wqt + (size_t)(n0 + r) * DIM + k0 + cc);
        }
    };

    v8f acc[4] = {};
    issue_tile(0, 0);
    const int T = DIM / KS;                      // 16 tiles
    for (int t = 0; t < T; ++t) {
        if (t + 1 < T) { issue_tile(t + 1, (t + 1) & 1); wait_async_le6(); }
        else          { wait_async_le0(); }
        __syncthreads();                         // tile t visible to all waves
        const int buf = t & 1;
#pragma unroll
        for (int kk = 0; kk < KS; kk += 32) {
            FragH a;
            a.q[0] = *reinterpret_cast<const uint4*>(&a_lds[buf][wave * 16 + row][kk + 16 * half]);
            a.q[1] = *reinterpret_cast<const uint4*>(&a_lds[buf][wave * 16 + row][kk + 16 * half + 8]);
#pragma unroll
            for (int ns = 0; ns < 4; ++ns) {
                FragH b;
                const uint4* bp = reinterpret_cast<const uint4*>(&b_lds[buf][ns * 16 + row][kk + 16 * half]);
                b.q[0] = bp[0];
                b.q[1] = bp[1];
                acc[ns] = __builtin_amdgcn_wmma_f32_16x16x32_f16(
                    false, a.v, false, b.v, (short)0, acc[ns], false, false);
            }
        }
        __syncthreads();                         // done reading buf before it is refilled
    }

#pragma unroll
    for (int ns = 0; ns < 4; ++ns) {
        const int n = n0 + ns * 16 + row;
#pragma unroll
        for (int r = 0; r < 8; ++r) {
            int m = m0 + wave * 16 + r + 8 * half;
            q[(size_t)m * DQ + n] = acc[ns][r];
        }
    }
}

// ---------------- BatchNorm stats: per-feature scale/shift ----------------
__global__ void bn_stats_kernel(const float* __restrict__ q, const float* __restrict__ gamma,
                                const float* __restrict__ beta,
                                float* __restrict__ bnA, float* __restrict__ bnB) {
    const int f = blockIdx.x, tid = threadIdx.x;
    float s = 0.f, ss = 0.f;
    for (int t = tid; t < BT; t += 256) {
        float v = q[(size_t)t * DQ + f];
        s += v; ss += v * v;
    }
    __shared__ float rs[256], rss[256];
    rs[tid] = s; rss[tid] = ss;
    __syncthreads();
    for (int o = 128; o > 0; o >>= 1) {
        if (tid < o) { rs[tid] += rs[tid + o]; rss[tid] += rss[tid + o]; }
        __syncthreads();
    }
    if (tid == 0) {
        float mean = rs[0] * (1.f / BT);
        float var  = rss[0] * (1.f / BT) - mean * mean;
        float a    = rsqrtf(var + BN_EPS) * gamma[f];
        bnA[f] = a;
        bnB[f] = beta[f] - mean * a;
    }
}

// ---------------- normalize + dots (WMMA) + 2-stage topk + softmax ----------------
// grid: (BT/16 token-tiles, HEADS), block 256 = 8 waves
__global__ void __launch_bounds__(256) dots_topk_kernel(
    const float* __restrict__ q, const _Float16* __restrict__ keysh,
    const float* __restrict__ bnA, const float* __restrict__ bnB,
    float* __restrict__ attnOut, int* __restrict__ idxOut) {
    const int tt = blockIdx.x;          // token tile
    const int h  = blockIdx.y;          // head
    const int tid = threadIdx.x, wave = tid >> 5, lane = tid & 31;

    __shared__ _Float16 qn[16][2][DH2];          // 8 KB
    __shared__ float    dotsS[2][16][NUM_KEYS];  // 32 KB
    __shared__ float    s2s[8][TOPK];
    __shared__ int      i2s[8][TOPK];

    // --- load + normalize q tile, convert to f16 ---
    for (int i = tid; i < 16 * 2 * DH2; i += 256) {
        int tokl = i >> 8;
        int rem  = i & 255;
        int p    = rem >> 7;
        int d    = rem & 127;
        int f    = p * 512 + h * DH2 + d;
        int tg   = tt * 16 + tokl;
        float v  = q[(size_t)tg * DQ + f] * bnA[f] + bnB[f];
        qn[tokl][p][d] = (_Float16)v;
    }
    __syncthreads();

    // --- phase 1: dots[p][tok][n] = qn · keys^T via WMMA ---
    const int row = lane & 15, half = lane >> 4;
    for (int job = wave; job < 32; job += 8) {
        int p = job >> 4, nt = job & 15, n0 = nt * 16;
        v8f c = {};
        const _Float16* bBase = keysh + ((size_t)(h * NUM_KEYS + n0 + row) * 2 + p) * DH2;
#pragma unroll
        for (int k0 = 0; k0 < DH2; k0 += 32) {
            FragH a, b;
#pragma unroll
            for (int v = 0; v < 8; ++v) {
                a.u[v] = *reinterpret_cast<const uint32_t*>(&qn[row][p][k0 + kOfA(v, half)]);
                b.u[v] = *reinterpret_cast<const uint32_t*>(bBase + k0 + kOfB(v, half));
            }
            c = __builtin_amdgcn_wmma_f32_16x16x32_f16(false, a.v, false, b.v, (short)0, c, false, false);
        }
        int n = n0 + row;
#pragma unroll
        for (int r = 0; r < 8; ++r) dotsS[p][r + 8 * half][n] = c[r];
    }
    __syncthreads();

    // --- phase 2: per-token two-stage top-k (wave handles 2 tokens) ---
    for (int ti = 0; ti < 2; ++ti) {
        const int tokl = wave * 2 + ti;
        const int tg   = tt * 16 + tokl;

        // stage A: top-32 of 256 per p (repeated wave argmax)
        float s_sel[2] = {0.f, 0.f};
        int   i_sel[2] = {0, 0};
#pragma unroll
        for (int p = 0; p < 2; ++p) {
            float vls[8];
#pragma unroll
            for (int e = 0; e < 8; ++e) vls[e] = dotsS[p][tokl][lane * 8 + e];
            unsigned taken = 0;
            for (int j = 0; j < TOPK; ++j) {
                float bv = -3.4e38f; int bn = 0x7fffffff;
#pragma unroll
                for (int e = 0; e < 8; ++e)
                    if (!((taken >> e) & 1) && vls[e] > bv) { bv = vls[e]; bn = lane * 8 + e; }
                for (int off = 16; off; off >>= 1) {
                    float ov = __shfl_xor(bv, off);
                    int   on = __shfl_xor(bn, off);
                    if (ov > bv || (ov == bv && on < bn)) { bv = ov; bn = on; }
                }
                if ((bn >> 3) == lane) taken |= 1u << (bn & 7);
                if (lane == j) { s_sel[p] = bv; i_sel[p] = bn; }
            }
        }

        // publish p=1 ranks to wave-private LDS scratch
        s2s[wave][lane] = s_sel[1];
        i2s[wave][lane] = i_sel[1];
        __syncthreads();   // uniform across block (ordering for intra-wave LDS reuse)

        // stage B: top-32 of the 32x32 cartesian sum; lane = p0 rank (row)
        const float s1 = s_sel[0];
        const int   i1 = i_sel[0];
        unsigned taken2 = 0;
        float fin_s = 0.f; int fin_idx = 0;
        for (int j = 0; j < TOPK; ++j) {
            float bv = -3.4e38f; int bn = 0x7fffffff;
            for (int cc = 0; cc < 32; ++cc)
                if (!((taken2 >> cc) & 1)) {
                    float cand = s1 + s2s[wave][cc];
                    if (cand > bv) { bv = cand; bn = lane * 32 + cc; }
                }
            for (int off = 16; off; off >>= 1) {
                float ov = __shfl_xor(bv, off);
                int   on = __shfl_xor(bn, off);
                if (ov > bv || (ov == bv && on < bn)) { bv = ov; bn = on; }
            }
            const int lw = bn >> 5, cw = bn & 31;
            int i1w  = __shfl(i1, lw);                       // uniform src lane
            int vidx = i1w * NUM_KEYS + i2s[wave][cw];
            if (lw == lane) taken2 |= 1u << cw;
            if (lane == j) { fin_s = bv; fin_idx = vidx; }
        }

        // softmax across the 32 selected (one per lane)
        float m = fin_s;
        for (int off = 16; off; off >>= 1) m = fmaxf(m, __shfl_xor(m, off));
        float e = __expf(fin_s - m);
        float sum = e;
        for (int off = 16; off; off >>= 1) sum += __shfl_xor(sum, off);
        float attn = e / sum;

        size_t base = ((size_t)tg * HEADS + h) * TOPK + lane;
        attnOut[base] = attn;
        idxOut[base]  = fin_idx;
        __syncthreads();   // before scratch reuse in next ti
    }
}

// ---------------- out[tok] = sum_j attn[j] * values[idx[j]] ----------------
// Dominant bandwidth phase: ~1 GB of gathered rows. Prefetch rows ahead.
__global__ void __launch_bounds__(256) gather_out_kernel(
    const float* __restrict__ attn, const int* __restrict__ vidx,
    const float* __restrict__ values, float* __restrict__ out) {
    const int tok = blockIdx.x, tid = threadIdx.x;
    __shared__ float sw[HEADS * TOPK];
    __shared__ int   si[HEADS * TOPK];
    if (tid < HEADS * TOPK) {
        sw[tid] = attn[(size_t)tok * HEADS * TOPK + tid];
        si[tid] = vidx[(size_t)tok * HEADS * TOPK + tid];
    }
    __syncthreads();
    const int d0 = tid * 4;
    // prime the pipeline: prefetch first few gathered rows
#pragma unroll
    for (int j = 0; j < 4; ++j)
        __builtin_prefetch(values + (size_t)si[j] * DIM + d0, 0, 1);
    float4 acc = make_float4(0.f, 0.f, 0.f, 0.f);
    for (int j = 0; j < HEADS * TOPK; ++j) {
        if (j + 4 < HEADS * TOPK)
            __builtin_prefetch(values + (size_t)si[j + 4] * DIM + d0, 0, 1);
        const float  w = sw[j];
        const float4 v = *reinterpret_cast<const float4*>(values + (size_t)si[j] * DIM + d0);
        acc.x += w * v.x; acc.y += w * v.y; acc.z += w * v.z; acc.w += w * v.w;
    }
    *reinterpret_cast<float4*>(out + (size_t)tok * DIM + d0) = acc;
}

extern "C" void kernel_launch(void* const* d_in, const int* in_sizes, int n_in,
                              void* d_out, int out_size, void* d_ws, size_t ws_size,
                              hipStream_t stream) {
    const float* x      = (const float*)d_in[0];
    const float* W_q    = (const float*)d_in[1];
    const float* keys   = (const float*)d_in[2];
    const float* values = (const float*)d_in[3];
    const float* gamma  = (const float*)d_in[4];
    const float* beta   = (const float*)d_in[5];
    float* out = (float*)d_out;

    char* ws = (char*)d_ws;
    _Float16* xh    = (_Float16*)ws;  ws += (size_t)BT * DIM * 2;
    _Float16* wqt   = (_Float16*)ws;  ws += (size_t)DIM * DQ * 2;
    _Float16* keysh = (_Float16*)ws;  ws += (size_t)HEADS * NUM_KEYS * 2 * DH2 * 2;
    float* qbuf     = (float*)ws;     ws += (size_t)BT * DQ * 4;
    float* bnA      = (float*)ws;     ws += (size_t)DQ * 4;
    float* bnB      = (float*)ws;     ws += (size_t)DQ * 4;
    float* attnBuf  = (float*)ws;     ws += (size_t)BT * HEADS * TOPK * 4;
    int*   idxBuf   = (int*)ws;       ws += (size_t)BT * HEADS * TOPK * 4;

    conv_f16_kernel<<<(BT * DIM) / 1024, 256, 0, stream>>>(x, xh, BT * DIM);
    transpose_wq_kernel<<<dim3(DQ / 32, DIM / 32), dim3(32, 8), 0, stream>>>(W_q, wqt);
    conv_f16_kernel<<<(HEADS * NUM_KEYS * 2 * DH2) / 1024, 256, 0, stream>>>(
        keys, keysh, HEADS * NUM_KEYS * 2 * DH2);
    gemm_q_kernel<<<dim3(DQ / 64, BT / 128), 256, 0, stream>>>(xh, wqt, qbuf);
    bn_stats_kernel<<<DQ, 256, 0, stream>>>(qbuf, gamma, beta, bnA, bnB);
    dots_topk_kernel<<<dim3(BT / 16, HEADS), 256, 0, stream>>>(
        qbuf, keysh, bnA, bnB, attnBuf, idxBuf);
    gather_out_kernel<<<BT, 256, 0, stream>>>(attnBuf, idxBuf, values, out);
}
